// aggregator_event_mtg_60988535603557
// MI455X (gfx1250) — matrix operand order repaired and checked by reading, not compiled, not run on hardware
//
#include <hip/hip_runtime.h>
#include <hip/hip_bf16.h>

// ---------------------------------------------------------------- constants
#define Hh   256
#define H2h  128
#define Ss   512
#define Gg   64
#define NNn  100000
#define NEe  200000
#define NENTn 20000
#define NRELn 500
#define Bb   32
#define SEQs 10

#define TM 128
#define TN 64
#define TK 32

typedef __attribute__((ext_vector_type(16))) __bf16 v16bf;
typedef __attribute__((ext_vector_type(8)))  float  v8f;
typedef __attribute__((ext_vector_type(2)))  float  v2f;
typedef __attribute__((ext_vector_type(2)))  __bf16 v2bf;

// one v_cvt_pk_bf16_f32 instead of manual rounding
__device__ __forceinline__ unsigned int pack2(float lo, float hi) {
  v2f f; f[0] = lo; f[1] = hi;
  v2bf b = __builtin_convertvector(f, v2bf);
  return __builtin_bit_cast(unsigned int, b);
}

__device__ __forceinline__ uint4 packA(float4 f0, float4 f1) {
  return make_uint4(pack2(f0.x, f0.y), pack2(f0.z, f0.w),
                    pack2(f1.x, f1.y), pack2(f1.z, f1.w));
}

// LDS tiles stored directly in WMMA fragment layout:
//   A (16x32 bf16 / 16-row tile): lane<16: M=lane, K={0..7,16..23}; lane>=16: M=lane-16, K={8..15,24..31}
//   B (32x16 / 16-col subtile):   lane: N=lane%16, K as A: lane<16 {0..7,16..23}, lane>=16 {8..15,24..31}
// Each lane's fragment = 16 contiguous bf16 = 32B -> two ds_load_b128.

#define WMMA_STEP(buf)                                                        \
  {                                                                           \
    v16bf afrag = *(const v16bf*)&As[buf][wave][lane][0];                     \
    v16bf bf0 = *(const v16bf*)&Bs[buf][0][lane][0];                          \
    v16bf bf1 = *(const v16bf*)&Bs[buf][1][lane][0];                          \
    v16bf bf2 = *(const v16bf*)&Bs[buf][2][lane][0];                          \
    v16bf bf3 = *(const v16bf*)&Bs[buf][3][lane][0];                          \
    acc[0] = __builtin_amdgcn_wmma_f32_16x16x32_bf16(false, afrag, false, bf0, (short)0, acc[0], false, false); \
    acc[1] = __builtin_amdgcn_wmma_f32_16x16x32_bf16(false, afrag, false, bf1, (short)0, acc[1], false, false); \
    acc[2] = __builtin_amdgcn_wmma_f32_16x16x32_bf16(false, afrag, false, bf2, (short)0, acc[2], false, false); \
    acc[3] = __builtin_amdgcn_wmma_f32_16x16x32_bf16(false, afrag, false, bf3, (short)0, acc[3], false, false); \
  }

// per-thread staging coordinates (A: 2 chunks, B: 1 half-fragment)
#define STAGE_COORDS()                                                        \
  const int rA0 = tid >> 2, rA1 = 64 + (tid >> 2), oA = tid & 3;              \
  const int tileA0 = rA0 >> 4, tileA1 = rA1 >> 4;                             \
  const int laneA0 = (rA0 & 15) + ((oA & 1) << 4);                            \
  const int laneA1 = (rA1 & 15) + ((oA & 1) << 4);                            \
  const int ebA = (oA >> 1) << 3;                                             \
  const int tB = tid >> 6, lB = (tid >> 1) & 31, eBs = tid & 1;               \
  const int nB = colBlock + ((tB << 4) | (lB & 15));                          \
  const int kbB = ((lB >> 4) << 3) + (eBs << 4);

#define LOAD_B(bR, Bptr, Nld, k0v)                                            \
  {                                                                           \
    const float* bp = &Bptr[(size_t)(k0v + kbB) * Nld + nB];                  \
    float b0 = bp[0], b1 = bp[Nld], b2 = bp[2 * Nld], b3 = bp[3 * Nld];       \
    float b4 = bp[4 * Nld], b5 = bp[5 * Nld], b6 = bp[6 * Nld], b7 = bp[7 * Nld]; \
    bR = make_uint4(pack2(b0, b1), pack2(b2, b3), pack2(b4, b5), pack2(b6, b7)); \
  }

#define STORE_STAGE(buf)                                                      \
  {                                                                           \
    *(uint4*)&As[buf][tileA0][laneA0][ebA] = aR0;                             \
    *(uint4*)&As[buf][tileA1][laneA1][ebA] = aR1;                             \
    *(uint4*)&Bs[buf][tB][lB][eBs << 3] = bR;                                 \
  }

// Double-buffered pipeline, unrolled by two K-steps (K must be multiple of 64)
// so buffer indices are compile-time 0/1 and LDS addresses are loop-invariant.
#define PIPELINE_LOOP(KTOT, LOADA, LOADBM, NLD)                               \
  LOADA(0);                                                                   \
  LOAD_B(bR, LOADBM, NLD, 0);                                                 \
  for (int k0 = 0; k0 < (KTOT); k0 += 2 * TK) {                               \
    STORE_STAGE(0);                                                           \
    __syncthreads();                                                          \
    LOADA(k0 + TK);                                                           \
    LOAD_B(bR, LOADBM, NLD, k0 + TK);                                         \
    WMMA_STEP(0);                                                             \
    STORE_STAGE(1);                                                           \
    __syncthreads();                                                          \
    if (k0 + 2 * TK < (KTOT)) {                                               \
      LOADA(k0 + 2 * TK);                                                     \
      LOAD_B(bR, LOADBM, NLD, k0 + 2 * TK);                                   \
    }                                                                         \
    WMMA_STEP(1);                                                             \
  }

// ================================================================
// Kernel 1: h0 = concat(ent_embeds[node_ids], ent_memory[node_ids]) @ W + b
// ================================================================
__global__ __launch_bounds__(256) void gemm_h0_kernel(
    const int* __restrict__ node_ids,
    const float* __restrict__ entE,   // [NENT, 512]
    const float* __restrict__ entM,   // [NENT, 256]
    const float* __restrict__ W,      // [768, 256]
    const float* __restrict__ bias,   // [256]
    float* __restrict__ C,            // [NN, 256]
    int M)
{
  __shared__ __align__(32) __bf16 As[2][8][32][16];
  __shared__ __align__(32) __bf16 Bs[2][4][32][16];
  __shared__ int nid[TM];
  const int tid = threadIdx.x;
  const int wave = tid >> 5, lane = tid & 31;
  const int half = lane >> 4, ln = lane & 15;
  const int rowBlock = blockIdx.x * TM;
  const int colBlock = blockIdx.y * TN;
  STAGE_COORDS();

  if (tid < TM) {
    int row = rowBlock + tid; if (row >= M) row = M - 1;
    nid[tid] = node_ids[row];
  }
  __syncthreads();

  // per-thread gather base pointers (uniform k0 selects entE vs entM)
  const float* apE0 = &entE[(size_t)nid[rA0] * 512 + (oA << 3)];
  const float* apM0 = &entM[(size_t)nid[rA0] * 256 + (oA << 3)];
  const float* apE1 = &entE[(size_t)nid[rA1] * 512 + (oA << 3)];
  const float* apM1 = &entM[(size_t)nid[rA1] * 256 + (oA << 3)];

  v8f acc[4];
  #pragma unroll
  for (int t = 0; t < 4; ++t)
    #pragma unroll
    for (int j = 0; j < 8; ++j) acc[t][j] = 0.f;

  uint4 aR0, aR1, bR;
  #define LOAD_A_H0(k0v)                                                      \
    {                                                                         \
      const float* p0 = ((k0v) < 512) ? apE0 + (k0v) : apM0 + ((k0v) - 512);  \
      const float* p1 = ((k0v) < 512) ? apE1 + (k0v) : apM1 + ((k0v) - 512);  \
      aR0 = packA(*(const float4*)p0, *(const float4*)(p0 + 4));              \
      aR1 = packA(*(const float4*)p1, *(const float4*)(p1 + 4));              \
    }
  PIPELINE_LOOP(768, LOAD_A_H0, W, 256)
  #undef LOAD_A_H0

  const int baseRow = rowBlock + wave * 16 + half * 8;
  if (rowBlock + TM <= M) {
    #pragma unroll
    for (int t = 0; t < 4; ++t) {
      int col = colBlock + t * 16 + ln;
      float bcol = bias[col];
      #pragma unroll
      for (int r = 0; r < 8; ++r) C[(size_t)(baseRow + r) * 256 + col] = acc[t][r] + bcol;
    }
  } else {
    #pragma unroll
    for (int t = 0; t < 4; ++t) {
      int col = colBlock + t * 16 + ln;
      #pragma unroll
      for (int r = 0; r < 8; ++r)
        if (baseRow + r < M) C[(size_t)(baseRow + r) * 256 + col] = acc[t][r] + bias[col];
    }
  }
}

// ================================================================
// Kernel 2: generic fused WMMA GEMM:
//   C = [relu]( A[M,K] @ B[K,N] (+ agg[row]/max(deg,1)) (+ bias) )
// K must be a multiple of 64 (true for all call sites: 256, 128)
// ================================================================
__global__ __launch_bounds__(256) void gemm_plain_kernel(
    const float* __restrict__ A, int lda,
    const float* __restrict__ Bm, int N,
    float* __restrict__ C,
    const float* __restrict__ bias,
    const float* __restrict__ agg,
    const float* __restrict__ deg,
    int relu, int M, int K)
{
  __shared__ __align__(32) __bf16 As[2][8][32][16];
  __shared__ __align__(32) __bf16 Bs[2][4][32][16];
  const int tid = threadIdx.x;
  const int wave = tid >> 5, lane = tid & 31;
  const int half = lane >> 4, ln = lane & 15;
  const int rowBlock = blockIdx.x * TM;
  const int colBlock = blockIdx.y * TN;
  STAGE_COORDS();

  int rowA0 = rowBlock + rA0; if (rowA0 >= M) rowA0 = M - 1;
  int rowA1 = rowBlock + rA1; if (rowA1 >= M) rowA1 = M - 1;
  const float* apA0 = &A[(size_t)rowA0 * lda + (oA << 3)];
  const float* apA1 = &A[(size_t)rowA1 * lda + (oA << 3)];

  v8f acc[4];
  #pragma unroll
  for (int t = 0; t < 4; ++t)
    #pragma unroll
    for (int j = 0; j < 8; ++j) acc[t][j] = 0.f;

  uint4 aR0, aR1, bR;
  #define LOAD_A_PLAIN(k0v)                                                   \
    {                                                                         \
      aR0 = packA(*(const float4*)(apA0 + (k0v)), *(const float4*)(apA0 + (k0v) + 4)); \
      aR1 = packA(*(const float4*)(apA1 + (k0v)), *(const float4*)(apA1 + (k0v) + 4)); \
    }
  PIPELINE_LOOP(K, LOAD_A_PLAIN, Bm, N)
  #undef LOAD_A_PLAIN

  const int baseRow = rowBlock + wave * 16 + half * 8;
  const bool full = (rowBlock + TM) <= M;
  float rdeg[8];
  #pragma unroll
  for (int r = 0; r < 8; ++r) rdeg[r] = 1.f;
  if (agg) {
    #pragma unroll
    for (int r = 0; r < 8; ++r) {
      int row = baseRow + r;
      float dg = deg[(full || row < M) ? row : (M - 1)];
      rdeg[r] = 1.f / (dg < 1.f ? 1.f : dg);
    }
  }
  #pragma unroll
  for (int t = 0; t < 4; ++t) {
    int col = colBlock + t * 16 + ln;
    float bcol = bias ? bias[col] : 0.f;
    if (full) {
      #pragma unroll
      for (int r = 0; r < 8; ++r) {
        int row = baseRow + r;
        float v = acc[t][r] + bcol;
        if (agg) v += agg[(size_t)row * N + col] * rdeg[r];
        if (relu && v < 0.f) v = 0.f;
        C[(size_t)row * N + col] = v;
      }
    } else {
      #pragma unroll
      for (int r = 0; r < 8; ++r) {
        int row = baseRow + r;
        if (row < M) {
          float v = acc[t][r] + bcol;
          if (agg) v += agg[(size_t)row * N + col] * rdeg[r];
          if (relu && v < 0.f) v = 0.f;
          C[(size_t)row * N + col] = v;
        }
      }
    }
  }
}

// ================================================================
// Kernel 3: message GEMM with fused scatter:
//   msg[e] = text[e]@Wt + hWn[src[e]] - eWn[type[e]];  atomicAdd agg[dst[e]]
// ================================================================
__global__ __launch_bounds__(256) void msg_kernel(
    const float* __restrict__ text,   // [NE, 512]
    const float* __restrict__ Wt,     // [512, N]
    const float* __restrict__ hWn,    // [NN, N]
    const float* __restrict__ eWn,    // [NREL, N]
    const int* __restrict__ src, const int* __restrict__ dst, const int* __restrict__ typ,
    float* __restrict__ agg,          // [NN, N]
    int N, int M)                     // M = NE
{
  __shared__ __align__(32) __bf16 As[2][8][32][16];
  __shared__ __align__(32) __bf16 Bs[2][4][32][16];
  __shared__ int esrc[TM], edst[TM], etyp[TM];
  const int tid = threadIdx.x;
  const int wave = tid >> 5, lane = tid & 31;
  const int half = lane >> 4, ln = lane & 15;
  const int rowBlock = blockIdx.x * TM;
  const int colBlock = blockIdx.y * TN;
  STAGE_COORDS();

  if (tid < TM) {
    int e = rowBlock + tid; if (e >= M) e = M - 1;
    esrc[tid] = src[e]; edst[tid] = dst[e]; etyp[tid] = typ[e];
  }

  int rowA0 = rowBlock + rA0; if (rowA0 >= M) rowA0 = M - 1;
  int rowA1 = rowBlock + rA1; if (rowA1 >= M) rowA1 = M - 1;
  const float* apA0 = &text[(size_t)rowA0 * Ss + (oA << 3)];
  const float* apA1 = &text[(size_t)rowA1 * Ss + (oA << 3)];

  v8f acc[4];
  #pragma unroll
  for (int t = 0; t < 4; ++t)
    #pragma unroll
    for (int j = 0; j < 8; ++j) acc[t][j] = 0.f;

  uint4 aR0, aR1, bR;
  #define LOAD_A_MSG(k0v)                                                     \
    {                                                                         \
      aR0 = packA(*(const float4*)(apA0 + (k0v)), *(const float4*)(apA0 + (k0v) + 4)); \
      aR1 = packA(*(const float4*)(apA1 + (k0v)), *(const float4*)(apA1 + (k0v) + 4)); \
    }
  PIPELINE_LOOP(Ss, LOAD_A_MSG, Wt, N)
  #undef LOAD_A_MSG

  const int baseLr = wave * 16 + half * 8;
  const bool full = (rowBlock + TM) <= M;
  #pragma unroll
  for (int t = 0; t < 4; ++t) {
    int col = colBlock + t * 16 + ln;
    if (full) {
      #pragma unroll
      for (int r = 0; r < 8; ++r) {
        int lr = baseLr + r;
        float v = acc[t][r] + hWn[(size_t)esrc[lr] * N + col] - eWn[(size_t)etyp[lr] * N + col];
        atomicAdd(&agg[(size_t)edst[lr] * N + col], v);
      }
    } else {
      #pragma unroll
      for (int r = 0; r < 8; ++r) {
        int lr = baseLr + r;
        if (rowBlock + lr < M) {
          float v = acc[t][r] + hWn[(size_t)esrc[lr] * N + col] - eWn[(size_t)etyp[lr] * N + col];
          atomicAdd(&agg[(size_t)edst[lr] * N + col], v);
        }
      }
    }
  }
}

// ================================================================
// Small GEMM (500-row relation path): one thread per output element.
// ================================================================
__global__ void small_gemm_kernel(
    const float* __restrict__ A0, const float* __restrict__ A1, int K0, int K,
    const float* __restrict__ Bm, const float* __restrict__ bias, int relu,
    float* __restrict__ C, int M, int N)
{
  int tid = blockIdx.x * blockDim.x + threadIdx.x;
  if (tid >= M * N) return;
  int m = tid / N, n = tid % N;
  float s = bias ? bias[n] : 0.f;
  const float* a0 = &A0[(size_t)m * K0];
  for (int k = 0; k < K0; ++k) s += a0[k] * Bm[k * N + n];
  if (K > K0) {
    const float* a1 = &A1[(size_t)m * (K - K0)];
    for (int k = 0; k < K - K0; ++k) s += a1[k] * Bm[(K0 + k) * N + n];
  }
  if (relu && s < 0.f) s = 0.f;
  C[tid] = s;
}

__global__ void deg_kernel(const int* __restrict__ dst, float* __restrict__ deg, int M) {
  int e = blockIdx.x * blockDim.x + threadIdx.x;
  if (e < M) atomicAdd(&deg[dst[e]], 1.f);
}

__global__ void pool_nodes_kernel(const float* __restrict__ h, const int* __restrict__ n2g,
                                  float* __restrict__ gn, int M) {
  int tid = blockIdx.x * blockDim.x + threadIdx.x;
  if (tid >= M * Hh) return;
  int n = tid >> 8, c = tid & 255;
  float v = h[tid];                 // post-relu, >= 0: int-ordered atomic max is valid
  atomicMax((int*)&gn[n2g[n] * Hh + c], __float_as_int(v));
}

__global__ void present_kernel(const int* __restrict__ e2g, const int* __restrict__ typ,
                               int* __restrict__ present, int M) {
  int e = blockIdx.x * blockDim.x + threadIdx.x;
  if (e < M) present[e2g[e] * NRELn + typ[e]] = 1;
}

__global__ void pool_edges_kernel(const int* __restrict__ present,
                                  const float* __restrict__ eType2,
                                  float* __restrict__ ge) {
  int tid = blockIdx.x * blockDim.x + threadIdx.x;
  if (tid >= Gg * Hh) return;
  int g = tid >> 8, c = tid & 255;
  float v = 0.f;
  for (int t = 0; t < NRELn; ++t)
    if (present[g * NRELn + t]) { float x = eType2[t * Hh + c]; if (x > v) v = x; }
  ge[tid] = v;
}

__global__ void output_kernel(const float* __restrict__ gn, const float* __restrict__ ge,
                              const int* __restrict__ time_idx, const int* __restrict__ len,
                              float* __restrict__ out) {
  int tid = blockIdx.x * blockDim.x + threadIdx.x;
  const int NSEQ = Bb * SEQs * (2 * Hh);
  if (tid < NSEQ) {
    int b = tid / (SEQs * 2 * Hh);
    int rem = tid % (SEQs * 2 * Hh);
    int s = rem / (2 * Hh), c = rem % (2 * Hh);
    float v = 0.f;
    if (s < len[b]) {
      int g = time_idx[b * SEQs + s];
      v = (c < Hh) ? gn[g * Hh + c] : ge[g * Hh + (c - Hh)];
    }
    out[tid] = v;
  } else if (tid < NSEQ + Bb) {
    out[tid] = (float)len[tid - NSEQ];
  }
}

// ================================================================
extern "C" void kernel_launch(void* const* d_in, const int* in_sizes, int n_in,
                              void* d_out, int out_size, void* d_ws, size_t ws_size,
                              hipStream_t stream) {
  const int*   node_ids  = (const int*)  d_in[0];
  const int*   edge_src  = (const int*)  d_in[1];
  const int*   edge_dst  = (const int*)  d_in[2];
  const int*   edge_type = (const int*)  d_in[3];
  const int*   node2g    = (const int*)  d_in[4];
  const int*   edge2g    = (const int*)  d_in[5];
  const float* text_emb  = (const float*)d_in[6];
  const int*   time_idx  = (const int*)  d_in[7];
  const int*   len_nz    = (const int*)  d_in[8];
  const float* ent_emb   = (const float*)d_in[9];
  const float* ent_mem   = (const float*)d_in[10];
  const float* rel_emb   = (const float*)d_in[11];
  const float* rel_mem   = (const float*)d_in[12];
  const float* w_node_W  = (const float*)d_in[13];
  const float* w_node_b  = (const float*)d_in[14];
  const float* w_rel_W   = (const float*)d_in[15];
  const float* w_rel_b   = (const float*)d_in[16];
  const float* Wn1 = (const float*)d_in[17];
  const float* Wl1 = (const float*)d_in[18];
  const float* Wt1 = (const float*)d_in[19];
  const float* Wr1 = (const float*)d_in[20];
  const float* b1  = (const float*)d_in[21];
  const float* Wn2 = (const float*)d_in[22];
  const float* Wl2 = (const float*)d_in[23];
  const float* Wt2 = (const float*)d_in[24];
  const float* Wr2 = (const float*)d_in[25];
  const float* b2  = (const float*)d_in[26];
  float* out = (float*)d_out;

  // --- workspace carving (bytes, 256-aligned) ---
  char* ws = (char*)d_ws;
  size_t off = 0;
  auto take = [&](size_t bytes) -> char* {
    char* p = ws + off; off += (bytes + 255) & ~(size_t)255; return p;
  };
  float* bufA  = (float*)take((size_t)NNn * Hh * 4);   // h0, later reused as h2
  float* h1    = (float*)take((size_t)NNn * H2h * 4);
  float* hWn   = (float*)take((size_t)NNn * Hh * 4);   // h@Wn (layer1: 128 cols, layer2: 256)
  float* agg   = (float*)take((size_t)NNn * Hh * 4);
  float* deg   = (float*)take((size_t)NNn * 4);
  float* eT0   = (float*)take((size_t)NRELn * Hh * 4);
  float* eWn1  = (float*)take((size_t)NRELn * H2h * 4);
  float* eT1   = (float*)take((size_t)NRELn * H2h * 4);
  float* eWn2  = (float*)take((size_t)NRELn * Hh * 4);
  float* eT2   = (float*)take((size_t)NRELn * Hh * 4);
  float* gn    = (float*)take((size_t)Gg * Hh * 4);
  float* ge    = (float*)take((size_t)Gg * Hh * 4);
  int*   pres  = (int*)  take((size_t)Gg * NRELn * 4);
  (void)ws_size;

  const int mblkN = (NNn + TM - 1) / TM;   // 782
  const int mblkE = (NEe + TM - 1) / TM;   // 1563

  // ---- relation-type path (500 rows, cheap VALU GEMMs) ----
  small_gemm_kernel<<<(NRELn * Hh + 255) / 256, 256, 0, stream>>>(
      rel_emb, rel_mem, 2 * Hh, 3 * Hh, w_rel_W, w_rel_b, 0, eT0, NRELn, Hh);
  small_gemm_kernel<<<(NRELn * H2h + 255) / 256, 256, 0, stream>>>(
      eT0, eT0, Hh, Hh, Wn1, nullptr, 0, eWn1, NRELn, H2h);
  small_gemm_kernel<<<(NRELn * H2h + 255) / 256, 256, 0, stream>>>(
      eT0, eT0, Hh, Hh, Wr1, nullptr, 1, eT1, NRELn, H2h);

  // ---- h0 = concat(gather) @ w_node_W + b ----
  gemm_h0_kernel<<<dim3(mblkN, Hh / TN), 256, 0, stream>>>(
      node_ids, ent_emb, ent_mem, w_node_W, w_node_b, bufA, NNn);

  // ---- layer 1 ----
  gemm_plain_kernel<<<dim3(mblkN, H2h / TN), 256, 0, stream>>>(
      bufA, Hh, Wn1, H2h, hWn, nullptr, nullptr, nullptr, 0, NNn, Hh);
  hipMemsetAsync(agg, 0, (size_t)NNn * H2h * 4, stream);
  hipMemsetAsync(deg, 0, (size_t)NNn * 4, stream);
  deg_kernel<<<(NEe + 255) / 256, 256, 0, stream>>>(edge_dst, deg, NEe);
  msg_kernel<<<dim3(mblkE, H2h / TN), 256, 0, stream>>>(
      text_emb, Wt1, hWn, eWn1, edge_src, edge_dst, edge_type, agg, H2h, NEe);
  gemm_plain_kernel<<<dim3(mblkN, H2h / TN), 256, 0, stream>>>(
      bufA, Hh, Wl1, H2h, h1, b1, agg, deg, 1, NNn, Hh);

  // ---- layer 2 relation path ----
  small_gemm_kernel<<<(NRELn * Hh + 255) / 256, 256, 0, stream>>>(
      eT1, eT1, H2h, H2h, Wn2, nullptr, 0, eWn2, NRELn, Hh);
  small_gemm_kernel<<<(NRELn * Hh + 255) / 256, 256, 0, stream>>>(
      eT1, eT1, H2h, H2h, Wr2, nullptr, 1, eT2, NRELn, Hh);

  // ---- layer 2 ----
  gemm_plain_kernel<<<dim3(mblkN, Hh / TN), 256, 0, stream>>>(
      h1, H2h, Wn2, Hh, hWn, nullptr, nullptr, nullptr, 0, NNn, H2h);
  hipMemsetAsync(agg, 0, (size_t)NNn * Hh * 4, stream);
  msg_kernel<<<dim3(mblkE, Hh / TN), 256, 0, stream>>>(
      text_emb, Wt2, hWn, eWn2, edge_src, edge_dst, edge_type, agg, Hh, NEe);
  gemm_plain_kernel<<<dim3(mblkN, Hh / TN), 256, 0, stream>>>(
      h1, H2h, Wl2, Hh, bufA /*h2*/, b2, agg, deg, 1, NNn, H2h);

  // ---- pooling ----
  hipMemsetAsync(gn, 0, (size_t)Gg * Hh * 4, stream);
  pool_nodes_kernel<<<((size_t)NNn * Hh + 255) / 256, 256, 0, stream>>>(bufA, node2g, gn, NNn);
  hipMemsetAsync(pres, 0, (size_t)Gg * NRELn * 4, stream);
  present_kernel<<<(NEe + 255) / 256, 256, 0, stream>>>(edge2g, edge_type, pres, NEe);
  pool_edges_kernel<<<(Gg * Hh + 255) / 256, 256, 0, stream>>>(pres, eT2, ge);

  // ---- ragged temporal gather + len output ----
  const int NOUT = Bb * SEQs * 2 * Hh + Bb;
  output_kernel<<<(NOUT + 255) / 256, 256, 0, stream>>>(gn, ge, time_idx, len_nz, out);
  (void)out_size; (void)n_in; (void)in_sizes;
}
